// BioNet_31791347925217
// MI455X (gfx1250) — matrix operand order repaired and checked by reading, not compiled, not run on hardware
//
#include <hip/hip_runtime.h>
#include <hip/hip_bf16.h>

// ---------------------------------------------------------------------------
// BioNet recurrence on MI455X (gfx1250):
//   X_{t+1} = MML( W @ X_t + X_bias ),  120 steps, W: 8192x8192, X: 8192x32
// Strategy: convert W to fp16 once (128 MB -> resident in 192 MB L2),
// run 120 chained WMMA kernels (v_wmma_f32_16x16x32_f16, f32 accumulation).
// Each block: 32-row x 32-col tile, K=8192 split over 8 waves (2048 waves
// total for occupancy), deterministic LDS reduction of partials.
// ---------------------------------------------------------------------------

#define NN 8192
#define BB 32
#define STEPS 120            // reference setup_inputs: max_steps = 120 (fixed)
#define KSPLIT 8             // waves per block, each owns NN/KSPLIT of K
#define KCHUNK (NN / KSPLIT) // 1024

typedef __attribute__((ext_vector_type(16))) _Float16 v16h;
typedef __attribute__((ext_vector_type(8)))  _Float16 v8h;
typedef __attribute__((ext_vector_type(8)))  float    v8f;

union HVec {
    v16h v;
    v8h  h[2];
};

// Michaelis-Menten-like activation (faithful to reference mml_activation).
__device__ __forceinline__ float mml(float x) {
    float fx = (x >= 0.0f) ? x : 0.01f * x;          // leaky relu, leak=0.01
    float shifted = 0.5f * (fx - 0.5f);
    bool  mask = (shifted < 0.0f);                   // fx < 0.5
    float gated = fx + (mask ? 10.0f : 0.0f);
    float right = 0.5f + shifted / gated;
    return mask ? fx : right;
}

// ---- one-time fp32 -> fp16 weight conversion (row-major preserved) --------
__global__ __launch_bounds__(256) void bionet_convert_w(
    const float* __restrict__ W, _Float16* __restrict__ Wh)
{
    const long long total = (long long)NN * NN;       // 64M elements
    long long stride = (long long)gridDim.x * blockDim.x * 8;
    for (long long i = ((long long)blockIdx.x * blockDim.x + threadIdx.x) * 8;
         i < total; i += stride) {
        const float4* p = (const float4*)(W + i);
        float4 a = p[0];
        float4 b = p[1];
        v8h h;
        h[0] = (_Float16)a.x; h[1] = (_Float16)a.y;
        h[2] = (_Float16)a.z; h[3] = (_Float16)a.w;
        h[4] = (_Float16)b.x; h[5] = (_Float16)b.y;
        h[6] = (_Float16)b.z; h[7] = (_Float16)b.w;
        *(v8h*)(Wh + i) = h;
    }
}

// ---- X_bias[n][b] = X_full[b][n] + bias[n]; zero-init X0 ------------------
__global__ __launch_bounds__(256) void bionet_prep(
    const float* __restrict__ Xfull, const float* __restrict__ bias,
    float* __restrict__ Xbias, _Float16* __restrict__ X0)
{
    int idx = blockIdx.x * blockDim.x + threadIdx.x;
    if (idx < NN * BB) {
        int n = idx >> 5;
        int b = idx & 31;
        Xbias[idx] = Xfull[(size_t)b * NN + n] + bias[n];
        X0[idx] = (_Float16)0.0f;
    }
}

// ---- one recurrence step: Xnext = MML(Wh @ Xcur + Xbias) ------------------
// Block = 8 waves. Block tile: rows [blk*32, blk*32+32) x all 32 batch cols.
// Wave w accumulates over K in [w*1024, (w+1)*1024); partials reduced in LDS
// in a fixed (deterministic) order, then bias + activation + fp16 store.
__global__ __launch_bounds__(256) void bionet_step(
    const _Float16* __restrict__ Wh,
    const _Float16* __restrict__ Xcur,
    _Float16*       __restrict__ Xnext,
    const float*    __restrict__ Xbias)
{
    __shared__ float red[KSPLIT * 4 * 8 * 32];        // 32 KB

    const int lane = threadIdx.x & 31;
    const int wave = threadIdx.x >> 5;                // 0..7 = K-slice owner
    const int m0   = blockIdx.x * 32;                 // block's first row

    // A-operand addressing per ISA 16-bit A 16x32 layout:
    //   lane m   (0..15): VGPR0-3 = K 0..7,  VGPR4-7 = K 16..23
    //   lane m+16       : VGPR0-3 = K 8..15, VGPR4-7 = K 24..31
    const int m     = lane & 15;
    const int half  = lane >> 4;
    const int kbase = wave * KCHUNK;

    const _Float16* wr0   = Wh + (size_t)(m0 + m) * NN + kbase + half * 8;
    const _Float16* wr1   = wr0 + (size_t)16 * NN;    // second M-tile
    // B-operand: lane l holds X column (l & 15), K rows in table order (stride BB).
    const _Float16* xbase = Xcur + (size_t)(kbase + half * 8) * BB + (lane & 15);

    v8f c00 = {};  // rows m0..+15,  cols 0..15
    v8f c01 = {};  // rows m0..+15,  cols 16..31
    v8f c10 = {};  // rows m0+16..31, cols 0..15
    v8f c11 = {};  // rows m0+16..31, cols 16..31

#pragma unroll 2
    for (int k0 = 0; k0 < KCHUNK; k0 += 32) {
        HVec a0, a1, b0, b1;
        a0.h[0] = *(const v8h*)(wr0 + k0);
        a0.h[1] = *(const v8h*)(wr0 + k0 + 16);
        a1.h[0] = *(const v8h*)(wr1 + k0);
        a1.h[1] = *(const v8h*)(wr1 + k0 + 16);
        const _Float16* xr = xbase + (size_t)k0 * BB;
#pragma unroll
        for (int s = 0; s < 8; ++s) {
            b0.v[s]     = xr[s * BB];
            b0.v[8 + s] = xr[(16 + s) * BB];
            b1.v[s]     = xr[s * BB + 16];
            b1.v[8 + s] = xr[(16 + s) * BB + 16];
        }
        c00 = __builtin_amdgcn_wmma_f32_16x16x32_f16(
                  false, a0.v, false, b0.v, (short)0, c00, false, false);
        c01 = __builtin_amdgcn_wmma_f32_16x16x32_f16(
                  false, a0.v, false, b1.v, (short)0, c01, false, false);
        c10 = __builtin_amdgcn_wmma_f32_16x16x32_f16(
                  false, a1.v, false, b0.v, (short)0, c10, false, false);
        c11 = __builtin_amdgcn_wmma_f32_16x16x32_f16(
                  false, a1.v, false, b1.v, (short)0, c11, false, false);
    }

    // ---- stage partials: red[((wave*4 + a)*8 + r)*32 + lane] --------------
    // (lane-consecutive addresses -> bank-conflict free)
    {
        v8f acc[4] = {c00, c01, c10, c11};
#pragma unroll
        for (int a = 0; a < 4; ++a)
#pragma unroll
            for (int r = 0; r < 8; ++r)
                red[((wave * 4 + a) * 8 + r) * 32 + lane] = acc[a][r];
    }
    __syncthreads();

    // ---- fixed-order reduce + bias + activation + store -------------------
    // Thread t owns (r = t>>5, l = t&31) for all 4 accumulators.
    {
        const int r = threadIdx.x >> 5;
        const int l = threadIdx.x & 31;
#pragma unroll
        for (int a = 0; a < 4; ++a) {
            float s = 0.0f;
#pragma unroll
            for (int w = 0; w < KSPLIT; ++w)
                s += red[((w * 4 + a) * 8 + r) * 32 + l];
            // C/D layout: VGPR r, lanes 0-15 -> M=r, N=l; lanes 16-31 -> M=r+8
            const int node  = m0 + (a >> 1) * 16 + r + 8 * (l >> 4);
            const int batch = (a & 1) * 16 + (l & 15);
            float x = s + Xbias[node * BB + batch];
            Xnext[node * BB + batch] = (_Float16)mml(x);
        }
    }
}

// ---- final: out[b][n] = (float) X[n][b] -----------------------------------
__global__ __launch_bounds__(256) void bionet_final(
    const _Float16* __restrict__ X, float* __restrict__ out)
{
    int idx = blockIdx.x * blockDim.x + threadIdx.x;   // idx = b*NN + n
    if (idx < NN * BB) {
        int n = idx & (NN - 1);
        int b = idx >> 13;
        out[idx] = (float)X[n * BB + b];
    }
}

extern "C" void kernel_launch(void* const* d_in, const int* in_sizes, int n_in,
                              void* d_out, int out_size, void* d_ws, size_t ws_size,
                              hipStream_t stream) {
    const float* X_full  = (const float*)d_in[0];   // [32, 8192]
    const float* weights = (const float*)d_in[1];   // [8192, 8192]
    const float* bias    = (const float*)d_in[2];   // [8192, 1]
    (void)in_sizes; (void)n_in; (void)out_size; (void)ws_size;

    // Workspace layout (requires ~130 MB):
    //   Wh    : fp16 weights          128 MB
    //   Xbias : f32 [N][B]              1 MB
    //   X0,X1 : fp16 ping-pong        0.5 MB each
    char* ws = (char*)d_ws;
    _Float16* Wh    = (_Float16*)(ws);
    float*    Xbias = (float*)   (ws + (size_t)NN * NN * 2);
    _Float16* X0    = (_Float16*)(ws + (size_t)NN * NN * 2 + (size_t)NN * BB * 4);
    _Float16* X1    = (_Float16*)(ws + (size_t)NN * NN * 2 + (size_t)NN * BB * 4
                                     + (size_t)NN * BB * 2);

    bionet_convert_w<<<4096, 256, 0, stream>>>(weights, Wh);
    bionet_prep<<<(NN * BB + 255) / 256, 256, 0, stream>>>(X_full, bias, Xbias, X0);

    _Float16* Xp[2] = {X0, X1};
    int cur = 0;
    for (int s = 0; s < STEPS; ++s) {
        bionet_step<<<NN / 32, 256, 0, stream>>>(Wh, Xp[cur], Xp[cur ^ 1], Xbias);
        cur ^= 1;
    }
    // STEPS is even -> final state is in X0
    bionet_final<<<(NN * BB + 255) / 256, 256, 0, stream>>>(Xp[cur], (float*)d_out);
}
